// Head_6571299963493
// MI455X (gfx1250) — matrix operand order repaired and checked by reading, not compiled, not run on hardware
//
#include <hip/hip_runtime.h>
#include <hip/hip_bf16.h>

typedef _Float16 v8h  __attribute__((ext_vector_type(8)));
typedef _Float16 v16h __attribute__((ext_vector_type(16)));
typedef float    v8f  __attribute__((ext_vector_type(8)));
typedef float    v4f  __attribute__((ext_vector_type(4)));
typedef int      vint4 __attribute__((vector_size(16)));   // matches builtin param type

#define BSZ   8
#define TLEN  2048
#define CEMB  1024
#define HD    128
#define TPAD  40   // padded row (halves) for transposed LDS tiles; 80B keeps 16B alignment

#if defined(__has_builtin)
#if __has_builtin(__builtin_amdgcn_global_load_async_to_lds_b128)
#define HAVE_ASYNC_LDS 1
#endif
#if __has_builtin(__builtin_amdgcn_s_wait_asynccnt)
#define HAVE_WAIT_ASYNC 1
#endif
#endif

static __device__ __forceinline__ v16h cat16(v8h g0, v8h g1) {
    v16h r;
#pragma unroll
    for (int i = 0; i < 8; ++i) { r[i] = g0[i]; r[i + 8] = g1[i]; }
    return r;
}

static __device__ __forceinline__ void wait_async0() {
#ifdef HAVE_WAIT_ASYNC
    __builtin_amdgcn_s_wait_asynccnt(0);
#else
    asm volatile("s_wait_asynccnt 0" ::: "memory");
#endif
}

// ---------------------------------------------------------------------------
// Phase 1: Q/K/V projections.  X[f32, B*T x 1024] @ W[f32, 1024 x 128] -> f16.
// Block tile: 64 rows x 128 cols, 8 waves = 4 row-tiles x 2 col-halves.
// Per 32-wide k-chunk both X and W are staged once in LDS (W transposed) so
// every WMMA operand is a contiguous 16B LDS read. grid.y selects Q/K/V.
// ---------------------------------------------------------------------------
__global__ __launch_bounds__(256)
void proj_kernel(const float* __restrict__ X,
                 const float* __restrict__ Wq,
                 const float* __restrict__ Wk,
                 const float* __restrict__ Wv,
                 _Float16* __restrict__ Qh,
                 _Float16* __restrict__ Kh,
                 _Float16* __restrict__ Vh)
{
    __shared__ __align__(16) _Float16 sX[64 * 32];      // [row][k]
    __shared__ __align__(16) _Float16 sWT[HD * TPAD];   // [n][k] transposed, padded

    const float* W  = (blockIdx.y == 0) ? Wq : (blockIdx.y == 1) ? Wk : Wv;
    _Float16*   Out = (blockIdx.y == 0) ? Qh : (blockIdx.y == 1) ? Kh : Vh;

    const int rb   = blockIdx.x * 64;     // block row base
    const int w    = threadIdx.x >> 5;
    const int lane = threadIdx.x & 31;
    const int hf   = lane >> 4;
    const int l16  = lane & 15;
    const int rt   = w >> 1;              // row tile 0..3
    const int ch   = w & 1;               // column half (64 cols)

    // cooperative-load thread mapping
    const int xrow = threadIdx.x >> 2;          // 0..63
    const int xkg  = (threadIdx.x & 3) * 8;     // k-group in chunk
    const int wkr  = threadIdx.x >> 3;          // 0..31  (k row of W chunk)
    const int wng  = (threadIdx.x & 7) * 16;    // n-group

    v8f acc[4];
#pragma unroll
    for (int nt = 0; nt < 4; ++nt) acc[nt] = {};

    for (int kc = 0; kc < CEMB / 32; ++kc) {
        // ---- global loads + f16 conversion (registers) ----
        const float* xp = X + (size_t)(rb + xrow) * CEMB + kc * 32 + xkg;
        v4f a0 = *(const v4f*)xp;
        v4f a1 = *(const v4f*)(xp + 4);
        v8h xh;
#pragma unroll
        for (int i = 0; i < 4; ++i) { xh[i] = (_Float16)a0[i]; xh[i + 4] = (_Float16)a1[i]; }

        const float* wp = W + (size_t)(kc * 32 + wkr) * HD + wng;
        v4f w0 = *(const v4f*)wp;
        v4f w1 = *(const v4f*)(wp + 4);
        v4f w2 = *(const v4f*)(wp + 8);
        v4f w3 = *(const v4f*)(wp + 12);
        _Float16 wh[16];
#pragma unroll
        for (int i = 0; i < 4; ++i) {
            wh[i]      = (_Float16)w0[i];
            wh[i + 4]  = (_Float16)w1[i];
            wh[i + 8]  = (_Float16)w2[i];
            wh[i + 12] = (_Float16)w3[i];
        }
        if (kc + 1 < CEMB / 32) {
            __builtin_prefetch(xp + 32, 0, 1);
            __builtin_prefetch(wp + 32 * HD, 0, 1);
        }

        __syncthreads();   // previous iteration fully consumed
        *(v8h*)&sX[xrow * 32 + xkg] = xh;
#pragma unroll
        for (int i = 0; i < 16; ++i)
            sWT[(wng + i) * TPAD + wkr] = wh[i];
        __syncthreads();

        // ---- compute: A from sX, B from sWT, 4 WMMAs per wave ----
        const _Float16* ap = &sX[(rt * 16 + l16) * 32 + hf * 8];
        v16h afr = cat16(*(const v8h*)ap, *(const v8h*)(ap + 16));
#pragma unroll
        for (int nt = 0; nt < 4; ++nt) {
            const _Float16* bp = &sWT[(ch * 64 + nt * 16 + l16) * TPAD + hf * 8];
            v16h bfr = cat16(*(const v8h*)bp, *(const v8h*)(bp + 16));
            acc[nt] = __builtin_amdgcn_wmma_f32_16x16x32_f16(false, afr, false, bfr,
                                                             (short)0, acc[nt],
                                                             false, false);
        }
    }

    // C layout: VGPR r -> row r + 8*hf, lane -> column
#pragma unroll
    for (int nt = 0; nt < 4; ++nt) {
        const int ocol = ch * 64 + nt * 16 + l16;
#pragma unroll
        for (int r = 0; r < 8; ++r) {
            const int orow = rb + rt * 16 + r + hf * 8;
            Out[(size_t)orow * HD + ocol] = (_Float16)acc[nt][r];
        }
    }
}

// ---------------------------------------------------------------------------
// Phase 2: causal flash attention over f16 Q/K/V, f32 output.
// Block = 8 waves = 128 query rows of one batch; 32 key columns per step.
// K tile staged via async global->LDS copies (ASYNCcnt path) when available.
// ---------------------------------------------------------------------------
__global__ __launch_bounds__(256)
void attn_kernel(const _Float16* __restrict__ Qh,
                 const _Float16* __restrict__ Kh,
                 const _Float16* __restrict__ Vh,
                 float* __restrict__ Out)
{
    __shared__ __align__(16) _Float16 sK[32 * HD];        // [s][h]
    __shared__ __align__(16) _Float16 sVT[HD * TPAD];     // [h][s] transposed
    __shared__ __align__(16) _Float16 sP[8][16 * 32];     // per-wave P scratch

    const int blk = blockIdx.x;            // 0..127
    const int b   = blk >> 4;              // batch
    const int rb  = (blk & 15) * 128;      // local query-row base
    const int w   = threadIdx.x >> 5;
    const int lane = threadIdx.x & 31;
    const int hf   = lane >> 4;
    const int l16  = lane & 15;
    const int m0   = rb + w * 16;
    const size_t base = (size_t)b * TLEN * HD;

    v16h qf[4];
    {
        const _Float16* qp = Qh + base + (size_t)(m0 + l16) * HD;
#pragma unroll
        for (int kc = 0; kc < 4; ++kc) {
            v8h g0 = *(const v8h*)(qp + kc * 32 + hf * 8);
            v8h g1 = *(const v8h*)(qp + kc * 32 + 16 + hf * 8);
            qf[kc] = cat16(g0, g1);
        }
    }

    v8f acc[8];
#pragma unroll
    for (int nt = 0; nt < 8; ++nt) acc[nt] = {};
    float mstat[8], lstat[8];
#pragma unroll
    for (int r = 0; r < 8; ++r) { mstat[r] = -__builtin_inff(); lstat[r] = 0.0f; }

    const float scale = 0.03125f;          // 1/sqrt(1024)
    const int jend = rb + 128;

    for (int jb = 0; jb < jend; jb += 32) {
        // ---- cooperative stage of K (async to LDS) and V (transposed) ----
        {
            const int srow = threadIdx.x >> 3;        // 0..31
            const int cg   = (threadIdx.x & 7) * 16;  // column group
            const _Float16* kp = Kh + base + (size_t)(jb + srow) * HD + cg;
            const _Float16* vp = Vh + base + (size_t)(jb + srow) * HD + cg;
#ifdef HAVE_ASYNC_LDS
            __builtin_amdgcn_global_load_async_to_lds_b128(
                (vint4*)kp,       (vint4*)&sK[srow * HD + cg],     0, 0);
            __builtin_amdgcn_global_load_async_to_lds_b128(
                (vint4*)(kp + 8), (vint4*)&sK[srow * HD + cg + 8], 0, 0);
#else
            v8h k0 = *(const v8h*)kp;
            v8h k1 = *(const v8h*)(kp + 8);
            *(v8h*)&sK[srow * HD + cg]     = k0;
            *(v8h*)&sK[srow * HD + cg + 8] = k1;
#endif
            v8h v0 = *(const v8h*)vp;
            v8h v1 = *(const v8h*)(vp + 8);
#pragma unroll
            for (int i = 0; i < 8; ++i) {
                sVT[(cg + i)     * TPAD + srow] = v0[i];
                sVT[(cg + 8 + i) * TPAD + srow] = v1[i];
            }
            if (jb + 32 < jend) {
                __builtin_prefetch(kp + 32 * HD, 0, 1);
                __builtin_prefetch(vp + 32 * HD, 0, 1);
            }
#ifdef HAVE_ASYNC_LDS
            wait_async0();
#endif
        }
        __syncthreads();

        if (jb <= m0 + 15) {
            // ---- S = Q K^T over 32 columns (two 16-wide tiles) ----
            v8f s0 = {}, s1 = {};
#pragma unroll
            for (int kc = 0; kc < 4; ++kc) {
                const _Float16* k0p = &sK[l16        * HD + kc * 32 + hf * 8];
                const _Float16* k1p = &sK[(16 + l16) * HD + kc * 32 + hf * 8];
                v16h b0 = cat16(*(const v8h*)k0p, *(const v8h*)(k0p + 16));
                v16h b1 = cat16(*(const v8h*)k1p, *(const v8h*)(k1p + 16));
                s0 = __builtin_amdgcn_wmma_f32_16x16x32_f16(false, qf[kc], false, b0,
                                                            (short)0, s0, false, false);
                s1 = __builtin_amdgcn_wmma_f32_16x16x32_f16(false, qf[kc], false, b1,
                                                            (short)0, s1, false, false);
            }
            // ---- scale + causal mask + online softmax ----
            float alpha[8];
#pragma unroll
            for (int r = 0; r < 8; ++r) {
                const int grow = m0 + r + hf * 8;
                s0[r] = ((jb + l16)      <= grow) ? s0[r] * scale : -__builtin_inff();
                s1[r] = ((jb + 16 + l16) <= grow) ? s1[r] * scale : -__builtin_inff();

                float tm = fmaxf(s0[r], s1[r]);
#pragma unroll
                for (int off = 1; off < 16; off <<= 1)
                    tm = fmaxf(tm, __shfl_xor(tm, off, 32));
                const float nm = fmaxf(mstat[r], tm);
                alpha[r] = __expf(mstat[r] - nm);
                mstat[r] = nm;
                const float p0 = __expf(s0[r] - nm);
                const float p1 = __expf(s1[r] - nm);
                s0[r] = p0; s1[r] = p1;
                float rs = p0 + p1;
#pragma unroll
                for (int off = 1; off < 16; off <<= 1)
                    rs += __shfl_xor(rs, off, 32);
                lstat[r] = lstat[r] * alpha[r] + rs;
            }
#pragma unroll
            for (int nt = 0; nt < 8; ++nt)
#pragma unroll
                for (int r = 0; r < 8; ++r)
                    acc[nt][r] *= alpha[r];

            // ---- P: C-layout -> A-layout via per-wave LDS ----
            _Float16* pw = &sP[w][0];
#pragma unroll
            for (int r = 0; r < 8; ++r) {
                const int prow = r + hf * 8;
                pw[prow * 32 + l16]      = (_Float16)s0[r];
                pw[prow * 32 + 16 + l16] = (_Float16)s1[r];
            }
            asm volatile("s_wait_dscnt 0" ::: "memory");
            v16h pf;
            {
                const _Float16* pp = pw + l16 * 32;
                pf = cat16(*(const v8h*)(pp + hf * 8),
                           *(const v8h*)(pp + 16 + hf * 8));
            }
            // ---- O += P V ----
#pragma unroll
            for (int nt = 0; nt < 8; ++nt) {
                const _Float16* vp = &sVT[(nt * 16 + l16) * TPAD + hf * 8];
                v16h vf = cat16(*(const v8h*)vp, *(const v8h*)(vp + 16));
                acc[nt] = __builtin_amdgcn_wmma_f32_16x16x32_f16(false, pf, false, vf,
                                                                 (short)0, acc[nt],
                                                                 false, false);
            }
        }
        __syncthreads();
    }

    // ---- normalize and store f32 output ----
#pragma unroll
    for (int r = 0; r < 8; ++r) {
        const float inv = 1.0f / lstat[r];
        const int grow = m0 + r + hf * 8;
        float* op = Out + ((size_t)b * TLEN + grow) * HD;
#pragma unroll
        for (int nt = 0; nt < 8; ++nt)
            op[nt * 16 + l16] = acc[nt][r] * inv;
    }
}

// ---------------------------------------------------------------------------
extern "C" void kernel_launch(void* const* d_in, const int* in_sizes, int n_in,
                              void* d_out, int out_size, void* d_ws, size_t ws_size,
                              hipStream_t stream)
{
    const float* X  = (const float*)d_in[0];
    const float* Wq = (const float*)d_in[1];
    const float* Wk = (const float*)d_in[2];
    const float* Wv = (const float*)d_in[3];
    float* Out = (float*)d_out;

    const size_t elems = (size_t)BSZ * TLEN * HD;
    _Float16* Qh = (_Float16*)d_ws;
    _Float16* Kh = Qh + elems;
    _Float16* Vh = Kh + elems;

    dim3 g1((BSZ * TLEN) / 64, 3);                  // 256 row-blocks x {Q,K,V}
    proj_kernel<<<g1, 256, 0, stream>>>(X, Wq, Wk, Wv, Qh, Kh, Vh);

    dim3 g2((BSZ * TLEN) / 128);                    // 128 blocks, 8 waves each
    attn_kernel<<<g2, 256, 0, stream>>>(Qh, Kh, Vh, Out);
}
